// Attention_2284922602251
// MI455X (gfx1250) — matrix operand order repaired and checked by reading, not compiled, not run on hardware
//
#include <hip/hip_runtime.h>

typedef __attribute__((ext_vector_type(16))) _Float16 v16h;
typedef __attribute__((ext_vector_type(8)))  float    v8f;

// ---- problem constants (fixed by reference setup) ----
constexpr int T   = 8192;
constexpr int H   = 16;
constexpr int HKV = 4;
constexpr int D   = 128;
constexpr int NB  = 64;
constexpr int BS  = 256;
constexpr int B   = 4;
constexpr int BPS = 8;
constexpr int S   = BPS * BS;      // 2048
constexpr int G   = H / HKV;       // 4
constexpr float SCALE  = 0.08838834764831845f;
constexpr float LOG2E  = 1.4426950408889634f;
constexpr float QSCALE = SCALE * LOG2E;   // softmax done in exp2 domain

constexpr int WAVES = 8;           // waves per workgroup
constexpr int BM    = WAVES * 16;  // 128 query rows per workgroup
constexpr int BN    = 32;          // keys per inner iteration (divides BS)

constexpr size_t CACHE_ELEMS = (size_t)NB * BS * HKV * D;   // 8,388,608

// ---------------------------------------------------------------------------
// kernel 1: build f16 working caches from the f32 input caches.
//   kh : [slot][hkv][d]          (row-major, for K B-fragments)
//   vt : [blk][hkv][d][kib]      (transposed per block, for V B-fragments)
// ---------------------------------------------------------------------------
__global__ void cache_build_kernel(const float* __restrict__ kc_in,
                                   const float* __restrict__ vc_in,
                                   _Float16* __restrict__ kh,
                                   _Float16* __restrict__ vt) {
  size_t e = (size_t)blockIdx.x * blockDim.x + threadIdx.x;
  if (e >= CACHE_ELEMS) return;
  kh[e] = (_Float16)kc_in[e];                       // identical linear layout
  int kib = (int)(e & (BS - 1));
  int d   = (int)((e >> 8) & (D - 1));
  int hkv = (int)((e >> 15) & (HKV - 1));
  int blk = (int)(e >> 17);
  vt[e] = (_Float16)vc_in[((size_t)(blk * BS + kib) * HKV + hkv) * D + d];
}

// ---------------------------------------------------------------------------
// kernel 2: paged KV scatter (slot < 0 dropped) into the f16 caches
// ---------------------------------------------------------------------------
__global__ void kv_scatter_kernel(const float* __restrict__ k,
                                  const float* __restrict__ v,
                                  const int* __restrict__ slot,
                                  _Float16* __restrict__ kh,
                                  _Float16* __restrict__ vt) {
  int i = blockIdx.x * blockDim.x + threadIdx.x;   // over T*HKV*D
  if (i >= T * HKV * D) return;
  int tok = i >> 9;                 // HKV*D == 512
  int rem = i & 511;
  int hkv = rem >> 7;
  int d   = rem & (D - 1);
  int s   = slot[tok];
  if (s < 0) return;
  kh[(size_t)s * (HKV * D) + rem] = (_Float16)k[i];
  int blk = s >> 8, kib = s & (BS - 1);
  vt[(((size_t)(blk * HKV + hkv) * D + d) << 8) + kib] = (_Float16)v[i];
}

// ---------------------------------------------------------------------------
// kernel 3: flash attention over the f16 paged caches
// ---------------------------------------------------------------------------
#define WMMA_F16(A, Bf, C) \
  __builtin_amdgcn_wmma_f32_16x16x32_f16(false, (A), false, (Bf), (short)0, (C), false, false)

// DPP lane permute (VALU; co-executes with the matrix pipe, keeps LDS pipe free)
template <int CTRL>
__device__ __forceinline__ float dppf(float x) {
  int xi = __builtin_bit_cast(int, x);
  int r  = __builtin_amdgcn_update_dpp(xi, xi, CTRL, 0xf, 0xf, true);
  return __builtin_bit_cast(float, r);
}
// max over each 16-lane group: quad swap, pair swap, half mirror, row mirror
__device__ __forceinline__ float rowmax16(float x) {
  x = fmaxf(x, dppf<0xB1>(x));    // quad_perm [1,0,3,2]
  x = fmaxf(x, dppf<0x4E>(x));    // quad_perm [2,3,0,1]
  x = fmaxf(x, dppf<0x141>(x));   // row_half_mirror
  x = fmaxf(x, dppf<0x140>(x));   // row_mirror
  return x;
}

// MODE 0: interior tile, no masking, both key sub-tiles
// MODE 1: diagonal tile, even wave (q0%32==0): sub0 masked (l>g*8+v), sub1 skipped
// MODE 2: diagonal tile, odd wave (q0%32==16): sub0 unmasked, sub1 masked (l>g*8+v)
template <int MODE>
__device__ __forceinline__ void attn_tile(
    const int g, const int l,
    const _Float16* __restrict__ kb_base, const _Float16* __restrict__ vb_base,
    const v16h (&qa)[4], const v16h& ones,
    v8f (&o)[8], v8f& o9, float (&rm)[8],
    _Float16 (&ptw)[16][BN]) {
  // ---- S = Q K^T ----
  v8f s0 = (v8f){}, s1 = (v8f){};
#pragma unroll
  for (int cc = 0; cc < 4; cc += 2) {
    v16h k00 = *(const v16h*)(kb_base + (size_t)l * (HKV * D) + cc * 32 + g * 16);
    v16h k10 = *(const v16h*)(kb_base + (size_t)l * (HKV * D) + (cc + 1) * 32 + g * 16);
    if (MODE != 1) {
      v16h k01 = *(const v16h*)(kb_base + (size_t)(16 + l) * (HKV * D) + cc * 32 + g * 16);
      v16h k11 = *(const v16h*)(kb_base + (size_t)(16 + l) * (HKV * D) + (cc + 1) * 32 + g * 16);
      s0 = WMMA_F16(qa[cc], k00, s0);
      s1 = WMMA_F16(qa[cc], k01, s1);
      s0 = WMMA_F16(qa[cc + 1], k10, s0);
      s1 = WMMA_F16(qa[cc + 1], k11, s1);
    } else {
      s0 = WMMA_F16(qa[cc], k00, s0);
      s0 = WMMA_F16(qa[cc + 1], k10, s0);
    }
  }

  // ---- row max via DPP butterfly (exp2 domain scores) ----
  float e0a[8], e1a[8], mx[8];
#pragma unroll
  for (int v = 0; v < 8; ++v) {
    float e0 = s0[v];
    float e1 = (MODE == 1) ? -1e30f : s1[v];
    if (MODE == 1 && (l > g * 8 + v)) e0 = -1e30f;   // static diagonal mask
    if (MODE == 2 && (l > g * 8 + v)) e1 = -1e30f;
    e0a[v] = e0;
    e1a[v] = e1;
    mx[v]  = rowmax16(fmaxf(e0, e1));
  }

  // ---- rescale O only when some running max actually moved (wave-uniform) ----
  bool bump = false;
#pragma unroll
  for (int v = 0; v < 8; ++v) bump |= (mx[v] > rm[v]);
  if (__any((int)bump)) {
#pragma unroll
    for (int v = 0; v < 8; ++v) {
      float mn = fmaxf(rm[v], mx[v]);
      float al = exp2f(rm[v] - mn);
      rm[v] = mn;
#pragma unroll
      for (int f = 0; f < 8; ++f) o[f][v] *= al;
      o9[v] *= al;
    }
  }

  // ---- P = exp2(S - rowmax): C layout -> LDS -> A layout ----
#pragma unroll
  for (int v = 0; v < 8; ++v) {
    ptw[g * 8 + v][l]      = (_Float16)exp2f(e0a[v] - rm[v]);
    ptw[g * 8 + v][16 + l] = (_Float16)exp2f(e1a[v] - rm[v]);
  }
  v16h pa;
#pragma unroll
  for (int i = 0; i < 8; ++i) {
    pa[i]     = ptw[l][g * 16 + i];
    pa[8 + i] = ptw[l][g * 16 + 8 + i];
  }

  // ---- row sums for free via a ones-column WMMA (no loads; runs under V loads) ----
  o9 = WMMA_F16(pa, ones, o9);

  // ---- O += P V : V fragment loads grouped 4-deep for MLP ----
#pragma unroll
  for (int ff = 0; ff < 8; ff += 4) {
    v16h vb0 = *(const v16h*)(vb_base + (((size_t)((ff + 0) * 16 + l)) << 8) + g * 16);
    v16h vb1 = *(const v16h*)(vb_base + (((size_t)((ff + 1) * 16 + l)) << 8) + g * 16);
    v16h vb2 = *(const v16h*)(vb_base + (((size_t)((ff + 2) * 16 + l)) << 8) + g * 16);
    v16h vb3 = *(const v16h*)(vb_base + (((size_t)((ff + 3) * 16 + l)) << 8) + g * 16);
    o[ff + 0] = WMMA_F16(pa, vb0, o[ff + 0]);
    o[ff + 1] = WMMA_F16(pa, vb1, o[ff + 1]);
    o[ff + 2] = WMMA_F16(pa, vb2, o[ff + 2]);
    o[ff + 3] = WMMA_F16(pa, vb3, o[ff + 3]);
  }
}

__global__ __launch_bounds__(WAVES * 32)
void paged_attn_kernel(const float* __restrict__ q,
                       const _Float16* __restrict__ kh,
                       const _Float16* __restrict__ vt,
                       const int*      __restrict__ bt,
                       float*          __restrict__ out) {
  const int mt   = blockIdx.x;
  const int h    = blockIdx.y;
  const int b    = blockIdx.z;
  const int hkv  = h / G;
  const int lane = threadIdx.x & 31;
  const int g    = lane >> 4;
  const int l    = lane & 15;
  // wave index / first query row, provably wave-uniform for the compiler:
  const int wave = __builtin_amdgcn_readfirstlane(threadIdx.x >> 5);
  const int q0   = mt * BM + wave * 16;

  __shared__ _Float16 pt[WAVES][16][BN];      // P re-layout staging (8 KB)

  // ---- Q A-fragments (SCALE * log2(e) folded into the f16 conversion) ----
  const float* qp = q + ((size_t)(b * S + q0 + l) * H + h) * D;
  v16h qa[4];
#pragma unroll
  for (int c = 0; c < 4; ++c) {
#pragma unroll
    for (int i = 0; i < 8; ++i) {
      qa[c][i]     = (_Float16)(qp[c * 32 + g * 16 + i] * QSCALE);
      qa[c][8 + i] = (_Float16)(qp[c * 32 + g * 16 + 8 + i] * QSCALE);
    }
  }

  v16h ones;
#pragma unroll
  for (int i = 0; i < 16; ++i) ones[i] = (_Float16)1.0f;

  v8f o[8];
#pragma unroll
  for (int f = 0; f < 8; ++f) o[f] = (v8f){};
  v8f o9 = (v8f){};                           // row sums (ones-column accumulator)
  float rm[8];
#pragma unroll
  for (int v = 0; v < 8; ++v) rm[v] = -1e30f;

  const int fl = q0 & ~(BN - 1);   // first diagonal tile; tiles below are mask-free

  for (int j = 0; j < fl; j += BN) {        // scalar trip count -> s_cbranch loop
    const int blk = bt[b * BPS + (j >> 8)];
    const int kib = j & (BS - 1);
    const _Float16* kb_base = kh + ((size_t)(blk * BS + kib) * HKV + hkv) * D;
    const _Float16* vb_base = vt + (((size_t)(blk * HKV + hkv) * D) << 8) + kib;
    if (((j + BN) & (BS - 1)) != 0) {
      __builtin_prefetch(kb_base + (size_t)(BN + l) * (HKV * D), 0, 1);
      __builtin_prefetch(vb_base + BN + ((size_t)(l * 8) << 8), 0, 1);
    }
    attn_tile<0>(g, l, kb_base, vb_base, qa, ones, o, o9, rm, pt[wave]);
  }

  {  // diagonal tile (scalar select)
    const int blk = bt[b * BPS + (fl >> 8)];
    const int kib = fl & (BS - 1);
    const _Float16* kb_base = kh + ((size_t)(blk * BS + kib) * HKV + hkv) * D;
    const _Float16* vb_base = vt + (((size_t)(blk * HKV + hkv) * D) << 8) + kib;
    if (q0 & 16) attn_tile<2>(g, l, kb_base, vb_base, qa, ones, o, o9, rm, pt[wave]);
    else         attn_tile<1>(g, l, kb_base, vb_base, qa, ones, o, o9, rm, pt[wave]);
  }

  // ---- normalize by the ones-column row sums and store ----
#pragma unroll
  for (int v = 0; v < 8; ++v) {
    const float inv = 1.0f / o9[v];
    float* op = out + ((size_t)(b * S + q0 + g * 8 + v) * H + h) * D;
#pragma unroll
    for (int f = 0; f < 8; ++f) op[f * 16 + l] = o[f][v] * inv;
  }
}

extern "C" void kernel_launch(void* const* d_in, const int* in_sizes, int n_in,
                              void* d_out, int out_size, void* d_ws, size_t ws_size,
                              hipStream_t stream) {
  const float* q     = (const float*)d_in[0];
  const float* k     = (const float*)d_in[1];
  const float* v     = (const float*)d_in[2];
  const float* kc_in = (const float*)d_in[3];
  const float* vc_in = (const float*)d_in[4];
  const int*   slot  = (const int*)d_in[5];
  const int*   bt    = (const int*)d_in[6];
  float*       out   = (float*)d_out;

  // f16 working caches in workspace (2 * 8.39M * 2B = 33.6 MB)
  _Float16* kh = (_Float16*)d_ws;
  _Float16* vt = kh + CACHE_ELEMS;

  {
    int nthreads = (int)CACHE_ELEMS;             // 8,388,608
    cache_build_kernel<<<(nthreads + 255) / 256, 256, 0, stream>>>(kc_in, vc_in, kh, vt);
  }
  {
    int nthreads = T * HKV * D;                  // 4,194,304
    kv_scatter_kernel<<<(nthreads + 255) / 256, 256, 0, stream>>>(k, v, slot, kh, vt);
  }
  dim3 grid(S / BM, H, B);
  paged_attn_kernel<<<grid, WAVES * 32, 0, stream>>>(q, kh, vt, bt, out);
}
